// NeuralStitcher_cross_att_50182397887019
// MI455X (gfx1250) — matrix operand order repaired and checked by reading, not compiled, not run on hardware
//
#include <hip/hip_runtime.h>
#include <hip/hip_bf16.h>
#include <math.h>

typedef __bf16 bf16;
typedef __attribute__((ext_vector_type(16))) __bf16 v16bf;
typedef __attribute__((ext_vector_type(8)))  __bf16 v8bf;
typedef __attribute__((ext_vector_type(8)))  float  v8f;

union AFrag { v16bf v; v8bf h[2]; };

#define B_  32
#define T_  128
#define N_  4000
#define R_  10
#define UD_ 32
#define RD_ 32
#define HD_ 16
#define H_  256
#define C_  8
#define D_  208
#define DP_ 224   // D padded to multiple of 32 for bf16 WMMA K-steps
#define NMAX_ 416 // per-region neuron capacity (actual = 400)

__device__ __forceinline__ float gelu_exact(float x) {
    return 0.5f * x * (1.0f + erff(x * 0.70710678118654752f));
}

__device__ __forceinline__ v8f wmma_bf16(v16bf a, v16bf b, v8f c) {
    // D(f32 16x16) = A(16x32 bf16) * B(32x16 bf16) + C
    return __builtin_amdgcn_wmma_f32_16x16x32_bf16(false, a, false, b, (short)0, c, false, false);
}

// ---------------------------------------------------------------------------
// Convert f32 weight [K][Nw] -> bf16 pre-swizzled WMMA-B fragment layout.
// dst element d: tile = d/512 (tile = kt*NT+nt), q = d%512, lane = q/16, e = q%16
// B layout (16-bit, 32x16): K_in_tile = (e&8)*2 + (lane>>4)*8 + (e&7), N_in_tile = lane&15
// => each lane's 16 B-frag elements are CONTIGUOUS in dst (two b128 loads).
// K rows >= K are zero-padded.
// ---------------------------------------------------------------------------
__global__ void swizzle_weight(const float* __restrict__ src, bf16* __restrict__ dst,
                               int K, int KT, int Nw) {
    int NT = Nw >> 4;
    int total = KT * NT * 512;
    for (int d = blockIdx.x * blockDim.x + threadIdx.x; d < total;
         d += gridDim.x * blockDim.x) {
        int tile = d >> 9, qq = d & 511;
        int lane = qq >> 4, e = qq & 15;
        int kt = tile / NT, nt = tile - kt * NT;
        int k = kt * 32 + ((e & 8) << 1) + ((lane >> 4) << 3) + (e & 7);
        int n = (nt << 4) + (lane & 15);
        float v = (k < K) ? src[(size_t)k * Nw + n] : 0.0f;
        dst[d] = (bf16)v;
    }
}

// ---------------------------------------------------------------------------
// Kernel A: feature concat + LN1 + K/V projections (bf16 WMMA), store bf16.
// grid (N/16, B), block 128 (4 waves). Each wave: 8 output tiles x 7 K-steps.
// ---------------------------------------------------------------------------
__global__ void __launch_bounds__(128)
kv_kernel(const float* __restrict__ x, const int* __restrict__ nr,
          const int* __restrict__ isl, const float* __restrict__ ue,
          const float* __restrict__ re, const float* __restrict__ he,
          const float* __restrict__ g1, const float* __restrict__ b1,
          const bf16* __restrict__ wk, const bf16* __restrict__ wv,
          bf16* __restrict__ kb, bf16* __restrict__ vb) {
    __shared__ float xc[16][DP_];
    __shared__ bf16  xbf[16][DP_];
    __shared__ float red[16][8][2];
    __shared__ float stats[16][2];
    int tid = threadIdx.x;
    int n0  = blockIdx.x * 16;
    int b   = blockIdx.y;

    // build concat features [16 neurons][208]
    for (int idx = tid; idx < 16 * D_; idx += 128) {
        int row = idx / D_, j = idx - row * D_;
        int n = n0 + row;
        float v;
        if (j < T_)                 v = x[((size_t)b * T_ + j) * N_ + n];
        else if (j < T_ + UD_)      v = ue[(size_t)n * UD_ + (j - T_)];
        else if (j < T_ + UD_ + RD_) v = re[(size_t)nr[n] * RD_ + (j - T_ - UD_)];
        else                        v = he[(size_t)isl[(size_t)b * N_ + n] * HD_ + (j - T_ - UD_ - RD_)];
        xc[row][j] = v;
    }
    __syncthreads();
    // LN1 stats: 8 partial slices per row
    {
        int row = tid >> 3, sl = tid & 7;
        float s = 0.f, s2 = 0.f;
        for (int j = sl; j < D_; j += 8) { float v = xc[row][j]; s += v; s2 += v * v; }
        red[row][sl][0] = s; red[row][sl][1] = s2;
    }
    __syncthreads();
    if (tid < 16) {
        float s = 0.f, s2 = 0.f;
        for (int i = 0; i < 8; ++i) { s += red[tid][i][0]; s2 += red[tid][i][1]; }
        float m = s / (float)D_;
        stats[tid][0] = m;
        stats[tid][1] = rsqrtf(s2 / (float)D_ - m * m + 1e-5f);
    }
    __syncthreads();
    // normalize -> bf16, zero-pad 208..223
    for (int idx = tid; idx < 16 * DP_; idx += 128) {
        int row = idx / DP_, j = idx - row * DP_;
        float v = 0.f;
        if (j < D_) v = (xc[row][j] - stats[row][0]) * stats[row][1] * g1[j] + b1[j];
        xbf[row][j] = (bf16)v;
    }
    __syncthreads();

    int wid = tid >> 5, lane = tid & 31;
    int mrow = lane & 15, g = lane >> 4;
    // hoist A fragments (K = 224 -> 7 steps); each = two aligned b128 LDS loads
    AFrag af[7];
    for (int kt = 0; kt < 7; ++kt) {
        const bf16* pr = &xbf[mrow][kt * 32 + 8 * g];
        af[kt].h[0] = *(const v8bf*)pr;
        af[kt].h[1] = *(const v8bf*)(pr + 16);
    }
    for (int sel = 0; sel < 2; ++sel) {
        const bf16* w    = sel ? wv : wk;
        bf16*       outp = sel ? vb : kb;
        for (int q4 = 0; q4 < 4; ++q4) {
            int nt = wid + q4 * 4;                  // 16 tiles over H=256
            v8f acc = {};
            for (int kt = 0; kt < 7; ++kt) {
                v16bf bf = *(const v16bf*)(w + ((size_t)(kt * 16 + nt) << 9) + lane * 16);
                acc = wmma_bf16(af[kt].v, bf, acc);
            }
            int col = nt * 16 + (lane & 15);
            for (int j = 0; j < 8; ++j) {
                int rowm = j + 8 * g;
                outp[((size_t)b * N_ + n0 + rowm) * H_ + col] = (bf16)acc[j];
            }
        }
    }
}

// ---------------------------------------------------------------------------
// Deterministic per-region neuron list (sequential single thread).
// ---------------------------------------------------------------------------
__global__ void region_prep(const int* __restrict__ nr, int* __restrict__ list,
                            int* __restrict__ cnt) {
    if (threadIdx.x == 0 && blockIdx.x == 0) {
        for (int r = 0; r < R_; ++r) cnt[r] = 0;
        for (int n = 0; n < N_; ++n) {
            int r = nr[n];
            if (r >= 0 && r < R_) { list[r * N_ + cnt[r]] = n; cnt[r]++; }
        }
    }
}

// ---------------------------------------------------------------------------
// Kernel B1: scores[b][n][c] = (q[c] . k[b][n]) / 16.  Wave per neuron.
// grid (N/8, B), block 256 (8 waves).
// ---------------------------------------------------------------------------
__global__ void __launch_bounds__(256)
score_kernel(const bf16* __restrict__ kb, const float* __restrict__ q,
             float* __restrict__ sc) {
    __shared__ float qs[C_][H_];
    int tid = threadIdx.x;
    for (int i = tid; i < C_ * H_; i += 256) qs[i / H_][i % H_] = q[i];
    __syncthreads();
    int b = blockIdx.y;
    int n = blockIdx.x * 8 + (tid >> 5);
    int lane = tid & 31;
    const bf16* kp = kb + ((size_t)b * N_ + n) * H_ + lane * 8;
    float kv[8];
    for (int i = 0; i < 8; ++i) kv[i] = (float)kp[i];
    float s[C_];
    for (int c = 0; c < C_; ++c) {
        float acc = 0.f;
        for (int i = 0; i < 8; ++i) acc += qs[c][lane * 8 + i] * kv[i];
        for (int off = 16; off; off >>= 1) acc += __shfl_xor(acc, off, 32);
        s[c] = acc * 0.0625f;  // 1/sqrt(H)
    }
    if (lane == 0) {
        float* op = sc + ((size_t)b * N_ + n) * C_;
        for (int c = 0; c < C_; ++c) op[c] = s[c];
    }
}

// ---------------------------------------------------------------------------
// Kernel B2: per-(b, region) masked softmax (all 8 channels) + attn@V + q residual.
// grid (R, B), block 256. Single pass over V with 8 register accumulators.
// ---------------------------------------------------------------------------
__global__ void __launch_bounds__(256)
attn_kernel(const float* __restrict__ sc, const bf16* __restrict__ vb,
            const int* __restrict__ list, const int* __restrict__ cnt,
            const float* __restrict__ q, float* __restrict__ latent) {
    __shared__ float W[C_][NMAX_];
    __shared__ int   nl[NMAX_];
    __shared__ float red[256];
    int r = blockIdx.x, b = blockIdx.y;
    int tid = threadIdx.x;
    int cn = cnt[r]; if (cn > NMAX_) cn = NMAX_;
    for (int i = tid; i < cn; i += 256) nl[i] = list[r * N_ + i];
    __syncthreads();
    for (int c = 0; c < C_; ++c) {
        float mx = -1e30f;
        for (int i = tid; i < cn; i += 256) {
            float s = sc[((size_t)b * N_ + nl[i]) * C_ + c];
            W[c][i] = s;
            mx = fmaxf(mx, s);
        }
        red[tid] = mx; __syncthreads();
        for (int off = 128; off; off >>= 1) {
            if (tid < off) red[tid] = fmaxf(red[tid], red[tid + off]);
            __syncthreads();
        }
        mx = red[0]; __syncthreads();
        float ls = 0.f;
        for (int i = tid; i < cn; i += 256) {
            float e = __expf(W[c][i] - mx); W[c][i] = e; ls += e;
        }
        red[tid] = ls; __syncthreads();
        for (int off = 128; off; off >>= 1) {
            if (tid < off) red[tid] += red[tid + off];
            __syncthreads();
        }
        float inv = 1.f / red[0]; __syncthreads();
        for (int i = tid; i < cn; i += 256) W[c][i] *= inv;
        __syncthreads();
    }
    // V read once; 8 channel accumulators per thread (thread == h)
    int h = tid;
    float acc[C_];
    for (int c = 0; c < C_; ++c) acc[c] = 0.f;
    for (int i = 0; i < cn; ++i) {
        float vv = (float)vb[((size_t)b * N_ + nl[i]) * H_ + h];
        for (int c = 0; c < C_; ++c) acc[c] += W[c][i] * vv;
    }
    for (int c = 0; c < C_; ++c)
        latent[(((size_t)b * R_ + r) * C_ + c) * H_ + h] = q[c * H_ + h] + acc[c];
}

// ---------------------------------------------------------------------------
// Kernel C: fused LN2 + MLP1(up/gelu/down + residual) + MLP2(ff1/gelu/ff2)
// + scatter to out[B,T,R,C].  16 latent rows per workgroup, 8 waves, bf16 WMMA.
// ---------------------------------------------------------------------------
__global__ void __launch_bounds__(256)
mlp_kernel(const float* __restrict__ latent,
           const float* __restrict__ g2, const float* __restrict__ b2,
           const bf16* __restrict__ wu,  const float* __restrict__ ub,
           const bf16* __restrict__ wd,  const float* __restrict__ db,
           const bf16* __restrict__ wf1, const float* __restrict__ f1b,
           const bf16* __restrict__ wf2, const float* __restrict__ f2b,
           float* __restrict__ out) {
    __shared__ float lat[16][H_];
    __shared__ bf16  xbf[16][H_];
    __shared__ bf16  hbf[16][512];
    __shared__ float red[16][16][2];
    __shared__ float stats[16][2];
    int tid = threadIdx.x;
    int g0 = blockIdx.x * 16;

    for (int idx = tid; idx < 16 * H_; idx += 256)
        lat[idx >> 8][idx & 255] = latent[(size_t)g0 * H_ + idx];
    __syncthreads();
    // LN2
    {
        int row = tid >> 4, sl = tid & 15;
        float s = 0.f, s2 = 0.f;
        for (int j = sl; j < H_; j += 16) { float v = lat[row][j]; s += v; s2 += v * v; }
        red[row][sl][0] = s; red[row][sl][1] = s2;
    }
    __syncthreads();
    if (tid < 16) {
        float s = 0.f, s2 = 0.f;
        for (int i = 0; i < 16; ++i) { s += red[tid][i][0]; s2 += red[tid][i][1]; }
        float m = s / 256.f;
        stats[tid][0] = m;
        stats[tid][1] = rsqrtf(s2 / 256.f - m * m + 1e-5f);
    }
    __syncthreads();
    for (int idx = tid; idx < 16 * H_; idx += 256) {
        int row = idx >> 8, j = idx & 255;
        xbf[row][j] = (bf16)((lat[row][j] - stats[row][0]) * stats[row][1] * g2[j] + b2[j]);
    }
    __syncthreads();

    int wid = tid >> 5, lane = tid & 31;
    int mrow = lane & 15, gg = lane >> 4;

    // GEMM1: hbf = gelu(xbf @ wu + ub)   [16x256]x[256x512]
    {
        AFrag af[8];
        for (int kt = 0; kt < 8; ++kt) {
            const bf16* pr = &xbf[mrow][kt * 32 + 8 * gg];
            af[kt].h[0] = *(const v8bf*)pr; af[kt].h[1] = *(const v8bf*)(pr + 16);
        }
        for (int q4 = 0; q4 < 4; ++q4) {
            int nt = wid + q4 * 8;
            v8f acc = {};
            for (int kt = 0; kt < 8; ++kt) {
                v16bf bf = *(const v16bf*)(wu + ((size_t)(kt * 32 + nt) << 9) + lane * 16);
                acc = wmma_bf16(af[kt].v, bf, acc);
            }
            int col = nt * 16 + (lane & 15);
            for (int j = 0; j < 8; ++j)
                hbf[j + 8 * gg][col] = (bf16)gelu_exact(acc[j] + ub[col]);
        }
    }
    __syncthreads();
    // GEMM2: lat += hbf @ wd + db; xbf = bf16(lat)   [16x512]x[512x256]
    for (int q2 = 0; q2 < 2; ++q2) {
        int nt = wid + q2 * 8;
        v8f acc = {};
        for (int kt = 0; kt < 16; ++kt) {
            AFrag a;
            const bf16* pr = &hbf[mrow][kt * 32 + 8 * gg];
            a.h[0] = *(const v8bf*)pr; a.h[1] = *(const v8bf*)(pr + 16);
            v16bf bf = *(const v16bf*)(wd + ((size_t)(kt * 16 + nt) << 9) + lane * 16);
            acc = wmma_bf16(a.v, bf, acc);
        }
        int col = nt * 16 + (lane & 15);
        for (int j = 0; j < 8; ++j) {
            int rowm = j + 8 * gg;
            float v = lat[rowm][col] + acc[j] + db[col];
            lat[rowm][col] = v;
            xbf[rowm][col] = (bf16)v;
        }
    }
    __syncthreads();
    // GEMM3: hbf = gelu(xbf @ wf1 + f1b)   [16x256]x[256x512]
    {
        AFrag af[8];
        for (int kt = 0; kt < 8; ++kt) {
            const bf16* pr = &xbf[mrow][kt * 32 + 8 * gg];
            af[kt].h[0] = *(const v8bf*)pr; af[kt].h[1] = *(const v8bf*)(pr + 16);
        }
        for (int q4 = 0; q4 < 4; ++q4) {
            int nt = wid + q4 * 8;
            v8f acc = {};
            for (int kt = 0; kt < 8; ++kt) {
                v16bf bf = *(const v16bf*)(wf1 + ((size_t)(kt * 32 + nt) << 9) + lane * 16);
                acc = wmma_bf16(af[kt].v, bf, acc);
            }
            int col = nt * 16 + (lane & 15);
            for (int j = 0; j < 8; ++j)
                hbf[j + 8 * gg][col] = (bf16)gelu_exact(acc[j] + f1b[col]);
        }
    }
    __syncthreads();
    // GEMM4: z = hbf @ wf2 + f2b -> out[b][t][r][c]   [16x512]x[512x128]
    {
        int nt = wid;                                   // 8 tiles over T=128
        v8f acc = {};
        for (int kt = 0; kt < 16; ++kt) {
            AFrag a;
            const bf16* pr = &hbf[mrow][kt * 32 + 8 * gg];
            a.h[0] = *(const v8bf*)pr; a.h[1] = *(const v8bf*)(pr + 16);
            v16bf bf = *(const v16bf*)(wf2 + ((size_t)(kt * 8 + nt) << 9) + lane * 16);
            acc = wmma_bf16(a.v, bf, acc);
        }
        int t = nt * 16 + (lane & 15);
        for (int j = 0; j < 8; ++j) {
            int grow = g0 + j + 8 * gg;                 // == ((b*R + r)*C + c)
            int bb = grow / (R_ * C_), rem = grow % (R_ * C_);
            int rr = rem >> 3, cc = rem & 7;
            out[(((size_t)bb * T_ + t) * R_ + rr) * C_ + cc] = acc[j] + f2b[t];
        }
    }
}

// ---------------------------------------------------------------------------
extern "C" void kernel_launch(void* const* d_in, const int* in_sizes, int n_in,
                              void* d_out, int out_size, void* d_ws, size_t ws_size,
                              hipStream_t stream) {
    const float* x   = (const float*)d_in[0];
    const int*   nr  = (const int*)d_in[1];
    const int*   isl = (const int*)d_in[2];
    const float* ue  = (const float*)d_in[3];
    const float* re  = (const float*)d_in[4];
    const float* he  = (const float*)d_in[5];
    const float* q   = (const float*)d_in[6];
    const float* kw  = (const float*)d_in[7];
    const float* vw  = (const float*)d_in[8];
    const float* g1  = (const float*)d_in[9];
    const float* b1  = (const float*)d_in[10];
    const float* g2  = (const float*)d_in[11];
    const float* b2  = (const float*)d_in[12];
    const float* uw  = (const float*)d_in[13];
    const float* ub  = (const float*)d_in[14];
    const float* dw  = (const float*)d_in[15];
    const float* db  = (const float*)d_in[16];
    const float* f1w = (const float*)d_in[17];
    const float* f1b = (const float*)d_in[18];
    const float* f2w = (const float*)d_in[19];
    const float* f2b = (const float*)d_in[20];
    float* out = (float*)d_out;

    char* ws = (char*)d_ws;
    size_t off = 0;
    auto alloc = [&](size_t bytes) {
        size_t a = (off + 255) & ~(size_t)255;
        off = a + bytes;
        return (void*)(ws + a);
    };
    bf16* kb   = (bf16*)alloc((size_t)B_ * N_ * H_ * sizeof(bf16));
    bf16* vb   = (bf16*)alloc((size_t)B_ * N_ * H_ * sizeof(bf16));
    float* sc  = (float*)alloc((size_t)B_ * N_ * C_ * sizeof(float));
    float* lat = (float*)alloc((size_t)B_ * R_ * C_ * H_ * sizeof(float));
    int* rlist = (int*)alloc((size_t)R_ * N_ * sizeof(int));
    int* rcnt  = (int*)alloc((size_t)R_ * sizeof(int));
    bf16* wkb  = (bf16*)alloc((size_t)DP_ * H_ * sizeof(bf16));    // 7x16 tiles
    bf16* wvb  = (bf16*)alloc((size_t)DP_ * H_ * sizeof(bf16));
    bf16* wub  = (bf16*)alloc((size_t)256 * 512 * sizeof(bf16));   // 8x32 tiles
    bf16* wdb  = (bf16*)alloc((size_t)512 * 256 * sizeof(bf16));   // 16x16 tiles
    bf16* wf1b = (bf16*)alloc((size_t)256 * 512 * sizeof(bf16));
    bf16* wf2b = (bf16*)alloc((size_t)512 * 128 * sizeof(bf16));   // 16x8 tiles

    auto gsz = [](int total) { return (total + 255) / 256; };
    swizzle_weight<<<gsz(7 * 16 * 512), 256, 0, stream>>>(kw,  wkb,  D_, 7,  H_);
    swizzle_weight<<<gsz(7 * 16 * 512), 256, 0, stream>>>(vw,  wvb,  D_, 7,  H_);
    swizzle_weight<<<gsz(8 * 32 * 512), 256, 0, stream>>>(uw,  wub,  256, 8,  512);
    swizzle_weight<<<gsz(16 * 16 * 512), 256, 0, stream>>>(dw,  wdb,  512, 16, 256);
    swizzle_weight<<<gsz(8 * 32 * 512), 256, 0, stream>>>(f1w, wf1b, 256, 8,  512);
    swizzle_weight<<<gsz(16 * 8 * 512), 256, 0, stream>>>(f2w, wf2b, 512, 16, 128);

    region_prep<<<1, 32, 0, stream>>>(nr, rlist, rcnt);

    kv_kernel<<<dim3(N_ / 16, B_), 128, 0, stream>>>(
        x, nr, isl, ue, re, he, g1, b1, wkb, wvb, kb, vb);

    score_kernel<<<dim3(N_ / 8, B_), 256, 0, stream>>>(kb, q, sc);

    attn_kernel<<<dim3(R_, B_), 256, 0, stream>>>(sc, vb, rlist, rcnt, q, lat);

    mlp_kernel<<<(B_ * R_ * C_) / 16, 256, 0, stream>>>(
        lat, g2, b2, wub, ub, wdb, db, wf1b, f1b, wf2b, f2b, out);
}
